// FullAttention_8340826489540
// MI455X (gfx1250) — matrix operand order repaired and checked by reading, not compiled, not run on hardware
//
#include <hip/hip_runtime.h>

// ---------------------------------------------------------------------------
// Causal flash attention, B=2, L=2048, H=16, D=64, fp32 in/out.
// f16 WMMA (v_wmma_f32_16x16x32_f16) + fp32 accum, wave32.
//
// Transposed formulation: S^T = K*Q^T and O^T = V^T*P, so the WMMA C-layout
// (lane = N) makes every softmax row lane-local. K/V chunks staged once per
// block in LDS as f16 (double-buffered) and consumed as A-operands; Q^T lives
// in registers as the B-operand with the softmax scale pre-folded. Causal
// masking only on the wave-uniform diagonal chunks; P->B fixup uses packed
// dword exchanges between lane halves.
// ---------------------------------------------------------------------------

typedef __attribute__((ext_vector_type(16))) _Float16 v16h;
typedef __attribute__((ext_vector_type(8)))  _Float16 v8h;
typedef __attribute__((ext_vector_type(8)))  float    v8f;
typedef __attribute__((ext_vector_type(4)))  float    v4f;

#define B_ 2
#define L_ 2048
#define H_ 16
#define D_ 64
#define ROWSTRIDE (H_ * D_)      // 1024 floats between consecutive seq positions
#define BLOCK_M 64               // queries per block (4 waves x 16)
#define CHUNK_N 32               // keys per iteration
#define KSTRIDE 72               // halfs per ldsK row [key][dim]  (144B, 16B-aligned, bank-skewed)
#define VSTRIDE 40               // halfs per ldsV row [dim][key]  (80B,  16B-aligned, bank-skewed)

__device__ __forceinline__ float fast_exp2(float x) {
#if __has_builtin(__builtin_amdgcn_exp2f)
  return __builtin_amdgcn_exp2f(x);
#else
  return exp2f(x);
#endif
}
__device__ __forceinline__ float fast_rcp(float x) {
#if __has_builtin(__builtin_amdgcn_rcpf)
  return __builtin_amdgcn_rcpf(x);
#else
  return 1.0f / x;
#endif
}

// A-operand load from LDS: lane row `p`, element e -> K=16g+e
__device__ __forceinline__ v16h ldA(const _Float16* p, int g) {
  v8h lo = *(const v8h*)(p + 16 * g);
  v8h hi = *(const v8h*)(p + 16 * g + 8);
  v16h r;
#pragma unroll
  for (int i = 0; i < 8; ++i) { r[i] = lo[i]; r[i + 8] = hi[i]; }
  return r;
}

// Cooperatively stage one 32-key chunk: K row-major f16 [key][dim],
// V transposed f16 [dim][key]. 128 threads.
__device__ __forceinline__ void stage_kv(const float* __restrict__ Kb,
                                         const float* __restrict__ Vb,
                                         _Float16* __restrict__ kbuf,
                                         _Float16* __restrict__ vbuf,
                                         int kbase, int t) {
  {   // K: thread t -> key = t>>2, dim quarter q = t&3 (dims 16q..16q+15)
    const int key = t >> 2, q = t & 3;
    const float* src = Kb + (size_t)(kbase + key) * ROWSTRIDE + 16 * q;
    v4f a = *(const v4f*)(src);
    v4f b = *(const v4f*)(src + 4);
    v4f c = *(const v4f*)(src + 8);
    v4f d = *(const v4f*)(src + 12);
    v8h lo, hi;
#pragma unroll
    for (int j = 0; j < 4; ++j) {
      lo[j]     = (_Float16)a[j];
      lo[j + 4] = (_Float16)b[j];
      hi[j]     = (_Float16)c[j];
      hi[j + 4] = (_Float16)d[j];
    }
    _Float16* dst = kbuf + key * KSTRIDE + 16 * q;   // 16B-aligned
    *(v8h*)dst       = lo;
    *(v8h*)(dst + 8) = hi;
  }
  {   // V^T: thread t -> key pair kp = t>>3, dim eighth q3 = t&7 (dims 8q3..+7)
    const int kp = t >> 3, q3 = t & 7;
    const float* r0 = Vb + (size_t)(kbase + 2 * kp) * ROWSTRIDE + 8 * q3;
    const float* r1 = r0 + ROWSTRIDE;
    v4f e0 = *(const v4f*)(r0);
    v4f e1 = *(const v4f*)(r0 + 4);
    v4f f0 = *(const v4f*)(r1);
    v4f f1 = *(const v4f*)(r1 + 4);
#pragma unroll
    for (int i = 0; i < 8; ++i) {
      const float ev = (i < 4) ? e0[i] : e1[i - 4];   // key 2kp
      const float ov = (i < 4) ? f0[i] : f1[i - 4];   // key 2kp+1
      union { _Float16 h[2]; unsigned int u; } p2;
      p2.h[0] = (_Float16)ev;
      p2.h[1] = (_Float16)ov;
      *(unsigned int*)(vbuf + (8 * q3 + i) * VSTRIDE + 2 * kp) = p2.u;  // b32
    }
  }
}

__global__ __launch_bounds__(128, 2)
void fa_causal_wmma_kernel(const float* __restrict__ Q, const float* __restrict__ K,
                           const float* __restrict__ V, float* __restrict__ O) {
  __shared__ _Float16 ldsK[2][CHUNK_N * KSTRIDE];   //  9216 B
  __shared__ _Float16 ldsV[2][D_ * VSTRIDE];        // 10240 B

  const int t    = threadIdx.x;
  const int lane = t & 31;
  const int wave = t >> 5;
  const int ln   = lane & 15;     // N column (query) inside a 16-wide tile
  const int g    = lane >> 4;     // half-wave group (selects key/dim subranges)

  const int bh = blockIdx.y;
  const int b  = bh >> 4;         // H_ == 16
  const int h  = bh & 15;
  const int qwave = blockIdx.x * BLOCK_M + wave * 16;
  const int qi    = qwave + ln;   // this lane's query row

  const size_t base = (size_t)b * L_ * ROWSTRIDE + (size_t)h * D_;
  const float* Qb = Q + base;
  const float* Kb = K + base;
  const float* Vb = V + base;
  float*       Ob = O + base;

  const float SC = 0.125f * 1.44269504088896341f;   // (1/sqrt(D)) * log2(e)

  // ---- Q^T as resident B-operand, pre-scaled by SC -------------------------
  // B layout: lane%16 = query col, element e -> dim = 16g + e (+32 for qb1)
  v16h qb0, qb1;
  {
    const float* qrow = Qb + (size_t)qi * ROWSTRIDE;
#pragma unroll
    for (int e = 0; e < 16; ++e) {
      qb0[e] = (_Float16)(qrow[16 * g + e] * SC);
      qb1[e] = (_Float16)(qrow[32 + 16 * g + e] * SC);
    }
  }

  // O^T accumulators: acc dc, reg r -> dim = 16*dc + 8g + r, query = qi
  v8f o0 = {}, o1 = {}, o2 = {}, o3 = {};
  float mrun = -__builtin_inff();
  float lrun = 0.0f;

  // Block-uniform causal bound; fully-masked chunks degenerate safely.
  const int nchunks = (blockIdx.x + 1) * (BLOCK_M / CHUNK_N);

  stage_kv(Kb, Vb, ldsK[0], ldsV[0], 0, t);
  __syncthreads();

  for (int c = 0; c < nchunks; ++c) {
    const int kbase = c * CHUNK_N;
    const _Float16* kbuf = ldsK[c & 1];
    const _Float16* vbuf = ldsV[c & 1];

    if (c + 1 < nchunks)
      stage_kv(Kb, Vb, ldsK[(c + 1) & 1], ldsV[(c + 1) & 1], kbase + CHUNK_N, t);
    if (c + 2 < nchunks) {
      __builtin_prefetch(Kb + (size_t)(kbase + 2 * CHUNK_N + (t >> 2)) * ROWSTRIDE + 16 * (t & 3), 0, 1);
      __builtin_prefetch(Vb + (size_t)(kbase + 2 * CHUNK_N + 2 * (t >> 3)) * ROWSTRIDE + 8 * (t & 7), 0, 1);
    }

    // ---- hoist all A-tiles: K tiles for S^T, V tiles for the later PV -----
    const _Float16* k0 = kbuf + ln * KSTRIDE;          // key kbase + ln
    const _Float16* k1 = kbuf + (16 + ln) * KSTRIDE;   // key kbase + 16 + ln
    v16h ka0 = ldA(k0, g);        // keys tile0, dims  0..31
    v16h ka1 = ldA(k0 + 32, g);   // keys tile0, dims 32..63
    v16h ka2 = ldA(k1, g);
    v16h ka3 = ldA(k1 + 32, g);
    v16h va0 = ldA(vbuf + (ln)      * VSTRIDE, g);     // dims  0..15 x keys
    v16h va1 = ldA(vbuf + (16 + ln) * VSTRIDE, g);
    v16h va2 = ldA(vbuf + (32 + ln) * VSTRIDE, g);
    v16h va3 = ldA(vbuf + (48 + ln) * VSTRIDE, g);

    // ---- S^T = K * (SC*Q)^T -----------------------------------------------
    v8f s0 = {}, s1 = {};
    s0 = __builtin_amdgcn_wmma_f32_16x16x32_f16(false, ka0, false, qb0, (short)0, s0, false, false);
    s0 = __builtin_amdgcn_wmma_f32_16x16x32_f16(false, ka1, false, qb1, (short)0, s0, false, false);
    s1 = __builtin_amdgcn_wmma_f32_16x16x32_f16(false, ka2, false, qb0, (short)0, s1, false, false);
    s1 = __builtin_amdgcn_wmma_f32_16x16x32_f16(false, ka3, false, qb1, (short)0, s1, false, false);

    // ---- lane-local softmax for query qi ----------------------------------
    // w[i], i<8: tile0 key = kbase+8g+i ; i>=8: tile1 key = kbase+16+8g+(i-8)
    float w[16];
    if (kbase + CHUNK_N - 1 <= qwave) {
      // wave-uniform fast path: every key in this chunk is visible to all lanes
#pragma unroll
      for (int r = 0; r < 8; ++r) { w[r] = s0[r]; w[r + 8] = s1[r]; }
    } else {
      const int dd = qi - kbase - 8 * g;   // key allowed iff local key idx <= dd
#pragma unroll
      for (int r = 0; r < 8; ++r) {
        w[r]     = (r <= dd)      ? s0[r] : -__builtin_inff();
        w[r + 8] = (r + 16 <= dd) ? s1[r] : -__builtin_inff();
      }
    }

    float m8[8];
#pragma unroll
    for (int i = 0; i < 8; ++i) m8[i] = fmaxf(w[i], w[i + 8]);
#pragma unroll
    for (int i = 0; i < 4; ++i) m8[i] = fmaxf(m8[i], m8[i + 4]);
    float vmax = fmaxf(fmaxf(m8[0], m8[1]), fmaxf(m8[2], m8[3]));
    vmax = fmaxf(vmax, __shfl_xor(vmax, 16, 32));   // combine the two key halves

    const float mnew  = fmaxf(mrun, vmax);
    const float alpha = fast_exp2(mrun - mnew);
    mrun = mnew;

    float p[16];
#pragma unroll
    for (int i = 0; i < 16; ++i) p[i] = fast_exp2(w[i] - mnew);

    float a8[8];
#pragma unroll
    for (int i = 0; i < 8; ++i) a8[i] = p[i] + p[i + 8];
#pragma unroll
    for (int i = 0; i < 4; ++i) a8[i] = a8[i] + a8[i + 4];
    float ssum = (a8[0] + a8[1]) + (a8[2] + a8[3]);
    ssum += __shfl_xor(ssum, 16, 32);
    lrun = lrun * alpha + ssum;

#pragma unroll
    for (int r = 0; r < 8; ++r) {
      o0[r] *= alpha; o1[r] *= alpha; o2[r] *= alpha; o3[r] *= alpha;
    }

    // ---- P -> B-layout (lane%16 = query, element e -> key 16g+e) ----------
    // Pack key pairs into dwords, exchange dwords between lane halves.
    v16h pb;
    {
      union { unsigned int u[8]; v16h v; } pbu;
#pragma unroll
      for (int j = 0; j < 4; ++j) {
        union { _Float16 h[2]; unsigned int u; } t0, t1;
        t0.h[0] = (_Float16)p[2 * j];        // tile0 keys kbase+8g+{2j,2j+1}
        t0.h[1] = (_Float16)p[2 * j + 1];
        t1.h[0] = (_Float16)p[8 + 2 * j];    // tile1 keys kbase+16+8g+{2j,2j+1}
        t1.h[1] = (_Float16)p[9 + 2 * j];
        const unsigned int r0 = (unsigned int)__shfl_xor((int)t0.u, 16, 32);
        const unsigned int r1 = (unsigned int)__shfl_xor((int)t1.u, 16, 32);
        pbu.u[j]     = g ? r1   : t0.u;   // keys 16g +     {2j,2j+1}
        pbu.u[4 + j] = g ? t1.u : r0;     // keys 16g + 8 + {2j,2j+1}
      }
      pb = pbu.v;
    }

    // ---- O^T += V^T * P ---------------------------------------------------
    o0 = __builtin_amdgcn_wmma_f32_16x16x32_f16(false, va0, false, pb, (short)0, o0, false, false);
    o1 = __builtin_amdgcn_wmma_f32_16x16x32_f16(false, va1, false, pb, (short)0, o1, false, false);
    o2 = __builtin_amdgcn_wmma_f32_16x16x32_f16(false, va2, false, pb, (short)0, o2, false, false);
    o3 = __builtin_amdgcn_wmma_f32_16x16x32_f16(false, va3, false, pb, (short)0, o3, false, false);

    __syncthreads();   // everyone done reading buf (c&1) before it is restaged
  }

  // ---- epilogue: normalize, swap halves so each lane owns 32 contiguous ----
  // dims of its query row, then wide coalesced stores.
  {
    const float inv = fast_rcp(lrun);
    float q0[8], q1[8], q2[8], q3[8];
#pragma unroll
    for (int r = 0; r < 8; ++r) {
      const float xa = g ? o0[r] : o2[r];
      const float xb = g ? o1[r] : o3[r];
      const float ra = __shfl_xor(xa, 16, 32);
      const float rb = __shfl_xor(xb, 16, 32);
      q0[r] = (g ? ra    : o0[r]) * inv;   // dims 32g + 0..7
      q1[r] = (g ? o2[r] : ra)    * inv;   // dims 32g + 8..15
      q2[r] = (g ? rb    : o1[r]) * inv;   // dims 32g + 16..23
      q3[r] = (g ? o3[r] : rb)    * inv;   // dims 32g + 24..31
    }
    float* dst = Ob + (size_t)qi * ROWSTRIDE + 32 * g;
#pragma unroll
    for (int r = 0; r < 8; ++r) {
      dst[r]      = q0[r];
      dst[8 + r]  = q1[r];
      dst[16 + r] = q2[r];
      dst[24 + r] = q3[r];
    }
  }
}

extern "C" void kernel_launch(void* const* d_in, const int* in_sizes, int n_in,
                              void* d_out, int out_size, void* d_ws, size_t ws_size,
                              hipStream_t stream) {
  (void)in_sizes; (void)n_in; (void)out_size; (void)d_ws; (void)ws_size;
  const float* Q = (const float*)d_in[0];
  const float* K = (const float*)d_in[1];
  const float* V = (const float*)d_in[2];
  // d_in[3] is the boolean causal mask; the kernel computes causality directly.
  float* O = (float*)d_out;

  dim3 grid(L_ / BLOCK_M, B_ * H_);   // (32, 32)
  dim3 block(128);                    // 4 waves, 16 query rows each
  fa_causal_wmma_kernel<<<grid, block, 0, stream>>>(Q, K, V, O);
}
